// TestModel_71751723647617
// MI455X (gfx1250) — compile-verified, matrix operand order checked by reading, not run on hardware
//
#include <hip/hip_runtime.h>
#include <math.h>

typedef __attribute__((ext_vector_type(16))) _Float16 v16h;
typedef __attribute__((ext_vector_type(8)))  _Float16 v8h;
typedef __attribute__((ext_vector_type(8)))  float    v8f;

#define Bsz 8
#define Ssz 512
#define Vsz 32000
#define Esz 32
#define HSsz 128
#define DIN 64   // E + H

// ---------------- kernel 1: wemb fp32 -> f16 (row-major V x E) ----------------
__global__ void cvt_wemb_f16(const float* __restrict__ w, _Float16* __restrict__ o, int n) {
    int i = blockIdx.x * blockDim.x + threadIdx.x;
    if (i < n) o[i] = (_Float16)w[i];
}

// ---------------- kernel 2: sequential recurrence, one block per batch --------
// 128 threads (4 waves). Lane t owns W1 row t (64 VGPRs). For the second layer,
// wave q (=t>>5) handles K-chunk q of W2 row j (=t&31), reduced with ds_add_f32.
__global__ __launch_bounds__(128, 1) void recurrence(
    const int*   __restrict__ tok,     // (B,S) token ids
    const float* __restrict__ h0,      // (B,1,H)
    const float* __restrict__ wemb,    // (V,E) fp32
    const float* __restrict__ W1,      // (HS, DIN)
    const float* __restrict__ b1,      // (HS)
    const float* __restrict__ W2,      // (E, HS)
    const float* __restrict__ b2,      // (E)
    _Float16*    __restrict__ statesH) // (B*S, E) f16, row-major (WMMA-A order)
{
    const int b = blockIdx.x;
    const int t = threadIdx.x;   // 0..127
    const int j = t & 31;
    const int q = t >> 5;

    __shared__ float d_sh[DIN];   // [x_t | h] broadcast-read
    __shared__ float z_sh[HSsz];
    __shared__ float hacc[Esz];

    // One-time weight staging into registers.
    float w1r[DIN];
    #pragma unroll
    for (int k = 0; k < DIN; ++k) w1r[k] = W1[t * DIN + k];
    const float b1i = b1[t];
    float w2q[32];
    #pragma unroll
    for (int k = 0; k < 32; ++k) w2q[k] = W2[j * HSsz + q * 32 + k];
    const float b2j = b2[j];

    if (t < Esz) {
        d_sh[t]       = wemb[(long)tok[b * Ssz + 0] * Esz + t];
        d_sh[Esz + t] = h0[b * Esz + t];
        hacc[t]       = b2j;
    }
    __syncthreads();

    for (int s = 0; s < Ssz; ++s) {
        // Phase A: z = gelu(W1 @ d + b1), exact gelu (erf form).
        float acc = b1i;
        #pragma unroll
        for (int k = 0; k < DIN; ++k) acc += w1r[k] * d_sh[k];   // broadcast reads
        z_sh[t] = 0.5f * acc * (1.0f + erff(acc * 0.70710678118654752f));
        __syncthreads();

        // Phase B: h = W2 @ z + b2 (bias pre-seeded in hacc), 4-way split over waves.
        float p = 0.0f;
        #pragma unroll
        for (int k = 0; k < 32; ++k) p += w2q[k] * z_sh[q * 32 + k]; // broadcast reads
        atomicAdd(&hacc[j], p);                                      // ds_add_f32
        __syncthreads();

        if (t < Esz) {
            float h = hacc[t];
            hacc[t]       = b2j;   // re-seed for next step
            d_sh[Esz + t] = h;     // recurrent feed
            statesH[((long)b * Ssz + s) * Esz + t] = (_Float16)h;
            if (s + 1 < Ssz)
                d_sh[t] = wemb[(long)tok[b * Ssz + s + 1] * Esz + t];
        }
        __syncthreads();
    }
}

// ---------------- kernel 3: out[m, v] = states[m,:] . wemb[v,:] + cls_b[v] ----
// M=4096, N=32000, K=32 == WMMA-f16 K exactly: one v_wmma per 16x16 tile.
// One wave handles one M-tile x 4 N-tiles (A reused across the 4 WMMAs).
__global__ __launch_bounds__(256) void cls_gemm(
    const _Float16* __restrict__ Ah,   // (4096, 32) f16 row-major
    const _Float16* __restrict__ Bh,   // (32000, 32) f16 row-major
    const float*    __restrict__ clsb, // (32000)
    float*          __restrict__ out)  // (4096, 32000)
{
    const int wave = (int)((blockIdx.x * blockDim.x + threadIdx.x) >> 5);
    const int lane = threadIdx.x & 31;
    const int lo   = lane & 15;
    const int hi   = lane >> 4;        // 0: K={0..7,16..23}, 1: K={8..15,24..31}

    const int NGRP  = Vsz / 64;        // 500 n-groups of 4 tiles
    const int mt    = wave / NGRP;     // 0..255
    const int ng    = wave % NGRP;     // 0..499
    const int m     = mt * 16 + lo;
    const int nbase = ng * 64;

    // A operand: per-lane 16 halves matching the 16-bit 16x32 A striping.
    const _Float16* ap = Ah + m * Esz + hi * 8;
    v8h alo = *(const v8h*)(ap);
    v8h ahi = *(const v8h*)(ap + 16);
    v16h a = __builtin_shufflevector(alo, ahi, 0,1,2,3,4,5,6,7,8,9,10,11,12,13,14,15);

    #pragma unroll
    for (int tt = 0; tt < 4; ++tt) {
        const int v = nbase + tt * 16 + lo;
        // B operand: column N=lo is wemb row v (K contiguous) -> mirrored striping.
        const _Float16* bp = Bh + v * Esz + hi * 8;
        v8h blo = *(const v8h*)(bp);
        v8h bhi = *(const v8h*)(bp + 16);
        v16h bb = __builtin_shufflevector(blo, bhi, 0,1,2,3,4,5,6,7,8,9,10,11,12,13,14,15);

        // Seed C with cls_b[v]: every C VGPR of this lane is column v -> free bias.
        float cb = clsb[v];
        v8f c = {cb, cb, cb, cb, cb, cb, cb, cb};
        c = __builtin_amdgcn_wmma_f32_16x16x32_f16(false, a, false, bb, (short)0, c, false, false);

        // Store 16x16 fp32 tile; stream past L2 (524 MB total output).
        float* op = out + (size_t)(mt * 16 + hi * 8) * Vsz + v;
        #pragma unroll
        for (int r = 0; r < 8; ++r)
            __builtin_nontemporal_store(c[r], op + (size_t)r * Vsz);
    }
}

extern "C" void kernel_launch(void* const* d_in, const int* in_sizes, int n_in,
                              void* d_out, int out_size, void* d_ws, size_t ws_size,
                              hipStream_t stream) {
    const int*   tok  = (const int*)d_in[0];
    const float* h0   = (const float*)d_in[1];
    const float* wemb = (const float*)d_in[2];
    const float* W1   = (const float*)d_in[3];
    const float* b1   = (const float*)d_in[4];
    const float* W2   = (const float*)d_in[5];
    const float* b2   = (const float*)d_in[6];
    const float* clsb = (const float*)d_in[7];
    float* out = (float*)d_out;

    _Float16* wembH   = (_Float16*)d_ws;                                  // 2,048,000 B
    _Float16* statesH = (_Float16*)((char*)d_ws + (size_t)Vsz * Esz * 2); // 262,144 B

    // 1) wemb -> f16
    cvt_wemb_f16<<<(Vsz * Esz + 255) / 256, 256, 0, stream>>>(wemb, wembH, Vsz * Esz);

    // 2) sequential scan, one block per batch element; emits f16 states
    recurrence<<<Bsz, 128, 0, stream>>>(tok, h0, wemb, W1, b1, W2, b2, statesH);

    // 3) WMMA classifier GEMM: 256 M-tiles x 500 n-groups = 128000 waves
    const int total_threads = 256 * 500 * 32;  // one wave per job
    cls_gemm<<<total_threads / 256, 256, 0, stream>>>(statesH, wembH, clsb, out);
}
